// Readout_81965155877094
// MI455X (gfx1250) — matrix operand order repaired and checked by reading, not compiled, not run on hardware
//
#include <hip/hip_runtime.h>
#include <hip/hip_bf16.h>

#define NN   400000
#define HH   256
#define CC   128
#define AA   32
#define GGH  512
#define NG   1024
#define EXT  (CC + AA)   // 160
#define HP   (HH + 8)    // padded LDS row (528B = 132 dwords -> bank skew of 4)

typedef __attribute__((ext_vector_type(16))) __bf16 v16bf;
typedef __attribute__((ext_vector_type(8)))  float  v8f;
typedef __attribute__((ext_vector_type(4)))  float  v4f;

// ---------------------------------------------------------------------------
// Pack a row-major f32 weight matrix W[K][Nn] into bf16 WMMA B-fragment order:
// frag index = ((kTile*nTiles + nTile)*32 + lane)*16 + i
//   kTile = k/32, kr = k%32, nTile = n/16
//   lane  = ((kr/8)%2 ? 16 : 0) + n%16,  i = kr%8 + (kr>=16 ? 8 : 0)
// so each lane's 16 bf16 (32B) are contiguous.
// ---------------------------------------------------------------------------
__global__ void pack_b_kernel(const float* __restrict__ W, __bf16* __restrict__ out,
                              int K, int Nn) {
    int idx = blockIdx.x * blockDim.x + threadIdx.x;
    if (idx >= K * Nn) return;
    int k = idx / Nn;
    int n = idx - k * Nn;
    int kTile = k >> 5, kr = k & 31;
    int nTile = n >> 4, nc = n & 15;
    int lane  = (((kr >> 3) & 1) << 4) + nc;
    int i     = (kr & 7) + ((kr >> 4) << 3);
    int nTiles = Nn >> 4;
    out[(((long)kTile * nTiles + nTile) * 32 + lane) * 16 + i] = (__bf16)W[idx];
}

__global__ void zero_f32_kernel(float* __restrict__ p, int n) {
    int i = blockIdx.x * blockDim.x + threadIdx.x;
    if (i < n) p[i] = 0.0f;
}

// ---------------------------------------------------------------------------
// Stage 1: nodewise gated readout + segment scatter-add.
// Block = 256 thr = 8 waves; block handles 64 node rows (4 M-tiles per wave
// for 4x B-fragment reuse); wave w handles columns [w*16, w*16+16) of C=128.
//   acc_g = [init|fin] @ Wg (K=512), acc_t = fin @ Wt (K=256)
//   out = sigmoid(acc_g + bg) * (acc_t + bt), atomic-added into readout[gid].
// Node states are streamed with NT hint (read-once, keep weights hot in L2).
// ---------------------------------------------------------------------------
__global__ __launch_bounds__(256) void stage1_kernel(
    const float* __restrict__ initS, const float* __restrict__ finS,
    const int* __restrict__ gids,
    const __bf16* __restrict__ WtP, const __bf16* __restrict__ WgP,
    const float* __restrict__ bg, const float* __restrict__ bt,
    float* __restrict__ readout)
{
    __shared__ __attribute__((aligned(16))) __bf16 sA_i[64 * HP];  // ~33 KB
    __shared__ __attribute__((aligned(16))) __bf16 sA_f[64 * HP];  // ~33 KB
    __shared__ int sg[64];

    const int tid = threadIdx.x;
    const long rowBase = (long)blockIdx.x * 64;

    // Cooperative staging: nontemporal f32x4 loads -> packed 4x bf16 -> b64 LDS store.
    for (int e = tid; e < 64 * (HH / 4); e += 256) {
        int r  = e >> 6;            // 64 quads per row
        int kq = (e & 63) << 2;
        v4f vi = __builtin_nontemporal_load((const v4f*)(initS + (rowBase + r) * HH + kq));
        v4f vf = __builtin_nontemporal_load((const v4f*)(finS  + (rowBase + r) * HH + kq));
        union { __bf16 h[4]; unsigned long long u8; } pi, pf;
        pi.h[0] = (__bf16)vi[0]; pi.h[1] = (__bf16)vi[1];
        pi.h[2] = (__bf16)vi[2]; pi.h[3] = (__bf16)vi[3];
        pf.h[0] = (__bf16)vf[0]; pf.h[1] = (__bf16)vf[1];
        pf.h[2] = (__bf16)vf[2]; pf.h[3] = (__bf16)vf[3];
        *(unsigned long long*)&sA_i[r * HP + kq] = pi.u8;
        *(unsigned long long*)&sA_f[r * HP + kq] = pf.u8;
    }
    if (tid < 64) sg[tid] = gids[rowBase + tid];
    __syncthreads();

    const int wave = tid >> 5;            // 0..7 -> column tile of C
    const int lane = tid & 31;
    const int row  = lane & 15;           // A-matrix M row within tile
    const int kOffA = (lane < 16) ? 0 : 8;

    v8f acc_t[4] = {{}, {}, {}, {}};
    v8f acc_g[4] = {{}, {}, {}, {}};

    for (int kb = 0; kb < HH; kb += 32) {
        const int kT = kb >> 5;
        v16bf b_t  = *(const v16bf*)(WtP + (((long)kT * 8 + wave) * 32 + lane) * 16);
        v16bf b_g0 = *(const v16bf*)(WgP + (((long)kT * 8 + wave) * 32 + lane) * 16);
        v16bf b_g1 = *(const v16bf*)(WgP + ((((long)kT + 8) * 8 + wave) * 32 + lane) * 16);
#pragma unroll
        for (int m = 0; m < 4; ++m) {
            const __bf16* pf = &sA_f[(m * 16 + row) * HP + kb + kOffA];
            const __bf16* pi = &sA_i[(m * 16 + row) * HP + kb + kOffA];
            v16bf a_f, a_i;
#pragma unroll
            for (int i = 0; i < 8; ++i) {
                a_f[i]     = pf[i];
                a_f[i + 8] = pf[16 + i];
                a_i[i]     = pi[i];
                a_i[i + 8] = pi[16 + i];
            }
            acc_t[m] = __builtin_amdgcn_wmma_f32_16x16x32_bf16(false, a_f, false, b_t,
                                                               (short)0, acc_t[m], false, false);
            acc_g[m] = __builtin_amdgcn_wmma_f32_16x16x32_bf16(false, a_i, false, b_g0,
                                                               (short)0, acc_g[m], false, false);
            acc_g[m] = __builtin_amdgcn_wmma_f32_16x16x32_bf16(false, a_f, false, b_g1,
                                                               (short)0, acc_g[m], false, false);
        }
    }

    const int n = wave * 16 + (lane & 15);
    const float bgv = bg[n];
    const float btv = bt[n];
    const int mBase = (lane < 16) ? 0 : 8;

#pragma unroll
    for (int m = 0; m < 4; ++m) {
        float vals[8];
#pragma unroll
        for (int r = 0; r < 8; ++r) {
            float gate = 1.0f / (1.0f + __expf(-(acc_g[m][r] + bgv)));
            vals[r] = gate * (acc_t[m][r] + btv);
        }
        // graph ids are sorted -> tiles are almost always single-graph: reduce first.
        if (sg[m * 16] == sg[m * 16 + 15]) {
            float sum = 0.0f;
#pragma unroll
            for (int r = 0; r < 8; ++r) sum += vals[r];
            sum += __shfl_xor(sum, 16);       // combine the two half-waves (same n)
            if (lane < 16)
                atomicAdd(&readout[(long)sg[m * 16] * CC + n], sum);
        } else {
#pragma unroll
            for (int r = 0; r < 8; ++r)
                atomicAdd(&readout[(long)sg[m * 16 + mBase + r] * CC + n], vals[r]);
        }
    }
}

// ---------------------------------------------------------------------------
// Stage 2: BatchNorm over NG=1024 rows of ext=[readout | aux] (160 cols),
// training-mode batch stats (biased variance). One block per column.
// Writes normalized bf16 row-major [NG][160].
// ---------------------------------------------------------------------------
__global__ __launch_bounds__(256) void bn_kernel(
    const float* __restrict__ readout, const float* __restrict__ aux,
    const float* __restrict__ gamma, const float* __restrict__ beta,
    __bf16* __restrict__ normB)
{
    const int col = blockIdx.x;      // 0..159
    const int tid = threadIdx.x;
    __shared__ float sred[256];
    __shared__ float smean, srstd;

    float x[4];
#pragma unroll
    for (int j = 0; j < 4; ++j) {
        int rowi = tid + j * 256;
        x[j] = (col < CC) ? readout[rowi * CC + col]
                          : aux[rowi * AA + (col - CC)];
    }
    float s = x[0] + x[1] + x[2] + x[3];
    sred[tid] = s;
    __syncthreads();
    for (int off = 128; off > 0; off >>= 1) {
        if (tid < off) sred[tid] += sred[tid + off];
        __syncthreads();
    }
    if (tid == 0) smean = sred[0] * (1.0f / NG);
    __syncthreads();
    const float mean = smean;

    float sq = 0.0f;
#pragma unroll
    for (int j = 0; j < 4; ++j) { float d = x[j] - mean; sq += d * d; }
    sred[tid] = sq;
    __syncthreads();
    for (int off = 128; off > 0; off >>= 1) {
        if (tid < off) sred[tid] += sred[tid + off];
        __syncthreads();
    }
    if (tid == 0) srstd = rsqrtf(sred[0] * (1.0f / NG) + 1e-5f);
    __syncthreads();

    const float scale = srstd * gamma[col];
    const float shift = beta[col];
#pragma unroll
    for (int j = 0; j < 4; ++j) {
        int rowi = tid + j * 256;
        normB[(long)rowi * EXT + col] = (__bf16)((x[j] - mean) * scale + shift);
    }
}

// ---------------------------------------------------------------------------
// Generic WMMA GEMM: D = act(A[M,K](bf16,row-major) @ Bpacked + bias).
// Block = 256 thr = 8 waves; block tile = 16 rows x 128 cols.
// grid.x = M/16, grid.y = Nn/128. Output either bf16 (hidden) or f32 (logits).
// ---------------------------------------------------------------------------
__global__ __launch_bounds__(256) void wmma_gemm_kernel(
    const __bf16* __restrict__ A, int lda,
    const __bf16* __restrict__ Bp, int K, int nTilesTotal,
    const float* __restrict__ bias,
    __bf16* __restrict__ outB, float* __restrict__ outF, int ldo, int relu)
{
    const int tid  = threadIdx.x;
    const int wave = tid >> 5;
    const int lane = tid & 31;
    const long rowBase = (long)blockIdx.x * 16;
    const int nTile = blockIdx.y * 8 + wave;
    const int n0 = nTile * 16;

    const int row = lane & 15;
    const int kOffA = (lane < 16) ? 0 : 8;

    v8f acc = {};
    for (int kb = 0; kb < K; kb += 32) {
        v16bf a;
        const __bf16* ap = A + (rowBase + row) * lda + kb + kOffA;
#pragma unroll
        for (int i = 0; i < 8; ++i) {
            a[i]     = ap[i];
            a[i + 8] = ap[16 + i];
        }
        v16bf b = *(const v16bf*)(Bp + (((long)(kb >> 5) * nTilesTotal + nTile) * 32 + lane) * 16);
        acc = __builtin_amdgcn_wmma_f32_16x16x32_bf16(false, a, false, b,
                                                      (short)0, acc, false, false);
    }

    const int n = n0 + (lane & 15);
    const float bv = bias[n];
    const int mBase = (lane < 16) ? 0 : 8;
#pragma unroll
    for (int r = 0; r < 8; ++r) {
        float v = acc[r] + bv;
        if (relu) v = fmaxf(v, 0.0f);
        long m = rowBase + mBase + r;
        if (outF) outF[m * ldo + n] = v;
        else      outB[m * ldo + n] = (__bf16)v;
    }
}

// ---------------------------------------------------------------------------
// Host-side orchestration
// ---------------------------------------------------------------------------
extern "C" void kernel_launch(void* const* d_in, const int* in_sizes, int n_in,
                              void* d_out, int out_size, void* d_ws, size_t ws_size,
                              hipStream_t stream) {
    const float* initS = (const float*)d_in[0];
    const float* finS  = (const float*)d_in[1];
    const float* aux   = (const float*)d_in[2];
    const int*   gids  = (const int*)d_in[3];
    const float* Wg    = (const float*)d_in[5];
    const float* bg    = (const float*)d_in[6];
    const float* Wt    = (const float*)d_in[7];
    const float* bt    = (const float*)d_in[8];
    const float* gamma = (const float*)d_in[9];
    const float* beta  = (const float*)d_in[10];
    const float* W1    = (const float*)d_in[11];
    const float* b1    = (const float*)d_in[12];
    const float* W2    = (const float*)d_in[13];
    const float* b2    = (const float*)d_in[14];
    float* out = (float*)d_out;

    char* ws = (char*)d_ws;
    size_t off = 0;
    auto alloc = [&](size_t bytes) { void* p = ws + off; off = (off + bytes + 255) & ~size_t(255); return p; };
    __bf16* WtP     = (__bf16*)alloc((size_t)HH * CC * 2);          // 64 KB
    __bf16* WgP     = (__bf16*)alloc((size_t)2 * HH * CC * 2);      // 128 KB
    __bf16* W1P     = (__bf16*)alloc((size_t)EXT * GGH * 2);        // 160 KB
    __bf16* W2P     = (__bf16*)alloc((size_t)GGH * CC * 2);         // 128 KB
    float*  readout = (float*) alloc((size_t)NG * CC * 4);          // 512 KB
    __bf16* normB   = (__bf16*)alloc((size_t)NG * EXT * 2);         // 320 KB
    __bf16* hiddenB = (__bf16*)alloc((size_t)NG * GGH * 2);         // 1 MB

    // Weight packing (bf16, WMMA B-fragment layout)
    pack_b_kernel<<<(HH * CC + 255) / 256, 256, 0, stream>>>(Wt, WtP, HH, CC);
    pack_b_kernel<<<(2 * HH * CC + 255) / 256, 256, 0, stream>>>(Wg, WgP, 2 * HH, CC);
    pack_b_kernel<<<(EXT * GGH + 255) / 256, 256, 0, stream>>>(W1, W1P, EXT, GGH);
    pack_b_kernel<<<(GGH * CC + 255) / 256, 256, 0, stream>>>(W2, W2P, GGH, CC);

    // Zero the scatter target (required every call: graph replays reuse d_ws)
    zero_f32_kernel<<<(NG * CC + 255) / 256, 256, 0, stream>>>(readout, NG * CC);

    // Stage 1: dominant GEMM + gated scatter-add (6250 blocks of 64 rows)
    stage1_kernel<<<NN / 64, 256, 0, stream>>>(initS, finS, gids, WtP, WgP, bg, bt, readout);

    // Stage 2: BatchNorm (batch stats) -> bf16 normalized ext
    bn_kernel<<<EXT, 256, 0, stream>>>(readout, aux, gamma, beta, normB);

    // Stage 3: MLP head via WMMA
    // hidden = relu(norm @ W1 + b1) : [1024,160] x [160,512]
    wmma_gemm_kernel<<<dim3(NG / 16, GGH / 128), 256, 0, stream>>>(
        normB, EXT, W1P, EXT, GGH / 16, b1, hiddenB, nullptr, GGH, 1);
    // logits = hidden @ W2 + b2 : [1024,512] x [512,128] -> f32 d_out
    wmma_gemm_kernel<<<dim3(NG / 16, CC / 128), 256, 0, stream>>>(
        hiddenB, GGH, W2P, GGH, CC / 16, b2, nullptr, out, CC, 0);
}
